// CGCNN_6270652252665
// MI455X (gfx1250) — compile-verified, matrix operand order checked by reading, not run on hardware
//
#include <hip/hip_runtime.h>
#include <hip/hip_bf16.h>
#include <cstdint>
#include <cstddef>

// ---------------------------------------------------------------------------
// CGCNN forward for MI455X (gfx1250, wave32, WMMA).
// Edge MLPs (2x [1.6M x 138] @ [138 x 64] x 3 layers ~ 170 GFLOP) run on
// v_wmma_f32_16x16x32_f16. A-fragments are gathered straight from global
// memory into the WMMA per-lane register layout (no LDS staging, no barrier);
// B-fragments live in LDS in pre-packed per-lane order, each read feeding 4
// WMMAs (2 M-tiles x {gate,cand}). Biases ride in as the WMMA C operand.
// Scatter uses HW global_atomic_add_f32 into a dummy-row-padded agg buffer.
// ---------------------------------------------------------------------------

typedef _Float16 v8h  __attribute__((ext_vector_type(8)));
typedef _Float16 v16h __attribute__((ext_vector_type(16)));
typedef float    v8f  __attribute__((ext_vector_type(8)));

#define HDIM 64
#define NCHUNK 5          // K = 138 padded to 5 chunks of 32

// Branch-free fast activations (native v_exp_f32 / v_log_f32).
__device__ __forceinline__ float softplusf(float x) {
    return fmaxf(x, 0.0f) + __logf(1.0f + __expf(-fabsf(x)));
}
__device__ __forceinline__ float sigmoidf(float x) {
    return __fdividef(1.0f, 1.0f + __expf(-x));
}

// ---------------------------------------------------------------------------
// h = node_features @ embed_w + embed_b ; also emit f16 copy for WMMA gathers
// ---------------------------------------------------------------------------
__global__ void embed_kernel(const float* __restrict__ nf,
                             const float* __restrict__ w,
                             const float* __restrict__ b,
                             float* __restrict__ h, _Float16* __restrict__ h16,
                             int Nn) {
    int idx = blockIdx.x * blockDim.x + threadIdx.x;
    if (idx >= Nn * HDIM) return;
    int i = idx >> 6, j = idx & 63;
    const float* row = nf + (size_t)i * 35;
    float acc = b[j];
#pragma unroll
    for (int k = 0; k < 35; ++k) acc += row[k] * w[k * HDIM + j];
    h[idx] = acc;
    h16[idx] = (_Float16)acc;
}

// ---------------------------------------------------------------------------
// Pack gate/cand weights [138,64] f32 into per-lane B-fragment order (f16):
// packed[((c*4+nt)*32+lane)*16+e] = W[k][n],
//   n = nt*16 + (lane&15), half = lane>>4,
//   k = c*32 + (e<8 ? half*8+e : 16+half*8+(e-8)),  k>=138 -> 0
// ---------------------------------------------------------------------------
__global__ void pack_weights(const float* __restrict__ gw,
                             const float* __restrict__ cw,
                             _Float16* __restrict__ pg,
                             _Float16* __restrict__ pc) {
    const int total = NCHUNK * 4 * 32 * 16; // 10240
    int idx = blockIdx.x * blockDim.x + threadIdx.x;
    if (idx >= 2 * total) return;
    int which = idx / total;
    int r = idx % total;
    int e = r & 15;
    int lane = (r >> 4) & 31;
    int nt = (r >> 9) & 3;
    int c = r >> 11;
    int hf = lane >> 4;
    int n = nt * 16 + (lane & 15);
    int k = c * 32 + ((e < 8) ? (hf * 8 + e) : (16 + hf * 8 + (e - 8)));
    float v = 0.0f;
    if (k < 138) v = which ? cw[k * HDIM + n] : gw[k * HDIM + n];
    (which ? pc : pg)[r] = (_Float16)v;
}

// Gather A-fragment chunks 0..3 (h_src K0-63, h_dst K64-127) for one M-tile.
__device__ __forceinline__ void loadHA(v16h* A, const _Float16* rs,
                                       const _Float16* rd, int hf) {
#pragma unroll
    for (int c = 0; c < 4; ++c) {
        const _Float16* row = (c < 2) ? rs : rd;
        v8h lo = *(const v8h*)(row + (c & 1) * 32 + hf * 8);
        v8h hi = *(const v8h*)(row + (c & 1) * 32 + 16 + hf * 8);
#pragma unroll
        for (int q = 0; q < 8; ++q) { A[c][q] = lo[q]; A[c][8 + q] = hi[q]; }
    }
}

// A-fragment chunk 4: K128-137 = edge features (f32->f16), K>=138 zero.
__device__ __forceinline__ v16h loadEfA(const float* __restrict__ ef, long e,
                                        int hf) {
    v16h a = {};
    const float2* p = (const float2*)(ef + (size_t)e * 10); // 8B-aligned (40B rows)
    if (hf == 0) {
#pragma unroll
        for (int q = 0; q < 4; ++q) {
            float2 v = p[q];
            a[2 * q] = (_Float16)v.x;
            a[2 * q + 1] = (_Float16)v.y;
        }
    } else {
        float2 v = p[4];
        a[0] = (_Float16)v.x;
        a[1] = (_Float16)v.y;
    }
    return a;
}

// ---------------------------------------------------------------------------
// Edge kernel: each wave handles 32 edges (two 16-row M-tiles) per iteration.
// ---------------------------------------------------------------------------
__global__ __launch_bounds__(256) void edge_conv(
    const _Float16* __restrict__ h16,
    const int* __restrict__ src, const int* __restrict__ dst,
    const float* __restrict__ ef,
    const _Float16* __restrict__ pg, const _Float16* __restrict__ pc,
    const float* __restrict__ gate_b, const float* __restrict__ cand_b,
    float* __restrict__ agg, int E, int Nn) {
    __shared__ __align__(32) _Float16 wg[NCHUNK * 4 * 32 * 16]; // 20 KB
    __shared__ __align__(32) _Float16 wc[NCHUNK * 4 * 32 * 16]; // 20 KB

    // Cooperative copy of packed weights into LDS (dword-wise).
    {
        const uint32_t* g32 = (const uint32_t*)pg;
        const uint32_t* c32 = (const uint32_t*)pc;
        uint32_t* wg32 = (uint32_t*)wg;
        uint32_t* wc32 = (uint32_t*)wc;
        for (int i = threadIdx.x; i < NCHUNK * 4 * 32 * 8; i += blockDim.x) {
            wg32[i] = g32[i];
            wc32[i] = c32[i];
        }
    }
    __syncthreads();

    const int wave = threadIdx.x >> 5;
    const int lane = threadIdx.x & 31;
    const int hf = lane >> 4;
    const int col = lane & 15;
    const int nT32 = (E + 31) >> 5;
    const int gw0 = blockIdx.x * (blockDim.x >> 5) + wave;
    const int wstride = gridDim.x * (blockDim.x >> 5);

    // Bias splats double as WMMA C operands (C[m][n] = b[n], column-per-lane).
    v8f gbias[4], cbias[4];
#pragma unroll
    for (int nt = 0; nt < 4; ++nt) {
        float gb = gate_b[nt * 16 + col];
        float cb = cand_b[nt * 16 + col];
#pragma unroll
        for (int i = 0; i < 8; ++i) { gbias[nt][i] = gb; cbias[nt][i] = cb; }
    }

    for (int t = gw0; t < nT32; t += wstride) {
        const long base = (long)t << 5;

        // ---- gather A fragments for two M-tiles straight from global ----
        v16h A0[NCHUNK], A1[NCHUNK];
        {
            const long e0 = base + col;
            const long e1 = e0 + 16;
            const long c0 = (e0 < E) ? e0 : 0;
            const long c1 = (e1 < E) ? e1 : 0;
            const int s0 = src[c0], d0 = dst[c0];
            const int s1 = src[c1], d1 = dst[c1];
            loadHA(A0, h16 + (size_t)s0 * HDIM, h16 + (size_t)d0 * HDIM, hf);
            loadHA(A1, h16 + (size_t)s1 * HDIM, h16 + (size_t)d1 * HDIM, hf);
            A0[4] = loadEfA(ef, c0, hf);
            A1[4] = loadEfA(ef, c1, hf);
        }

        // ---- scatter byte offsets (invalid edges -> dummy row Nn) ----
        uint32_t off0[8], off1[8];
#pragma unroll
        for (int i = 0; i < 8; ++i) {
            long e = base + 8 * hf + i;
            int r0 = (e < E) ? src[e] : Nn;
            int r1 = (e + 16 < E) ? src[e + 16] : Nn;
            off0[i] = (uint32_t)r0 * (HDIM * 4) + (uint32_t)col * 4;
            off1[i] = (uint32_t)r1 * (HDIM * 4) + (uint32_t)col * 4;
        }

        // ---- 4 N-tiles x 5 K-chunks x 2 M-tiles x {gate,cand} = 80 WMMA ----
#pragma unroll
        for (int nt = 0; nt < 4; ++nt) {
            v8f ag0 = gbias[nt], ac0 = cbias[nt];
            v8f ag1 = gbias[nt], ac1 = cbias[nt];
#pragma unroll
            for (int c = 0; c < NCHUNK; ++c) {
                const v16h bg = *(const v16h*)&wg[((c * 4 + nt) * 32 + lane) * 16];
                const v16h bc = *(const v16h*)&wc[((c * 4 + nt) * 32 + lane) * 16];
                ag0 = __builtin_amdgcn_wmma_f32_16x16x32_f16(
                    false, A0[c], false, bg, (short)0, ag0, false, false);
                ag1 = __builtin_amdgcn_wmma_f32_16x16x32_f16(
                    false, A1[c], false, bg, (short)0, ag1, false, false);
                ac0 = __builtin_amdgcn_wmma_f32_16x16x32_f16(
                    false, A0[c], false, bc, (short)0, ac0, false, false);
                ac1 = __builtin_amdgcn_wmma_f32_16x16x32_f16(
                    false, A1[c], false, bc, (short)0, ac1, false, false);
            }
#pragma unroll
            for (int i = 0; i < 8; ++i) {
                float v0 = sigmoidf(ag0[i]) * softplusf(ac0[i]);
                float v1 = sigmoidf(ag1[i]) * softplusf(ac1[i]);
                unsafeAtomicAdd((float*)((char*)agg + off0[i] + nt * 64), v0);
                unsafeAtomicAdd((float*)((char*)agg + off1[i] + nt * 64), v1);
            }
        }
    }
}

// ---------------------------------------------------------------------------
// BatchNorm stats: per-channel sum and sum of squares over N rows.
// stats[0..63]=sum, stats[64..127]=sumsq
// ---------------------------------------------------------------------------
__global__ __launch_bounds__(256) void bn_stats(const float* __restrict__ agg,
                                                float* __restrict__ stats, int Nn) {
    __shared__ float ls[256], ls2[256];
    const int c = threadIdx.x & 63;
    const int grp = threadIdx.x >> 6;
    float s = 0.0f, s2 = 0.0f;
    for (int r = blockIdx.x * 4 + grp; r < Nn; r += gridDim.x * 4) {
        float v = agg[(size_t)r * HDIM + c];
        s += v;
        s2 += v * v;
    }
    ls[threadIdx.x] = s;
    ls2[threadIdx.x] = s2;
    __syncthreads();
    if (threadIdx.x < 64) {
        s = ls[c] + ls[c + 64] + ls[c + 128] + ls[c + 192];
        s2 = ls2[c] + ls2[c + 64] + ls2[c + 128] + ls2[c + 192];
        unsafeAtomicAdd(&stats[c], s);
        unsafeAtomicAdd(&stats[64 + c], s2);
    }
}

// ---------------------------------------------------------------------------
// h = softplus(h + (agg-mu)*rsqrt(var+eps)*gamma + beta) ; refresh f16 copy
// ---------------------------------------------------------------------------
__global__ void bn_update(float* __restrict__ h, _Float16* __restrict__ h16,
                          const float* __restrict__ agg,
                          const float* __restrict__ stats,
                          const float* __restrict__ gamma,
                          const float* __restrict__ beta, int Nn) {
    int idx = blockIdx.x * blockDim.x + threadIdx.x;
    if (idx >= Nn * HDIM) return;
    int c = idx & 63;
    float invN = 1.0f / (float)Nn;
    float mu = stats[c] * invN;
    float var = stats[64 + c] * invN - mu * mu;
    float x = h[idx] + (agg[idx] - mu) * rsqrtf(var + 1e-5f) * gamma[c] + beta[c];
    float y = softplusf(x);
    h[idx] = y;
    h16[idx] = (_Float16)y;
}

// ---------------------------------------------------------------------------
// Per-graph mean pool via atomics.
// ---------------------------------------------------------------------------
__global__ void pool_kernel(const float* __restrict__ h,
                            const int* __restrict__ gidx,
                            float* __restrict__ pooled, float* __restrict__ cnts,
                            int Nn) {
    int idx = blockIdx.x * blockDim.x + threadIdx.x;
    if (idx >= Nn * HDIM) return;
    int i = idx >> 6, c = idx & 63;
    int g = gidx[i];
    unsafeAtomicAdd(&pooled[(size_t)g * HDIM + c], h[idx]);
    if (c == 0) unsafeAtomicAdd(&cnts[g], 1.0f);
}

// ---------------------------------------------------------------------------
// Final MLP: 64 -> 128 -> 64 -> 1 per graph (one block each).
// ---------------------------------------------------------------------------
__global__ __launch_bounds__(128) void mlp_kernel(
    const float* __restrict__ pooled, const float* __restrict__ cnts,
    const float* __restrict__ w1, const float* __restrict__ b1,
    const float* __restrict__ w2, const float* __restrict__ b2,
    const float* __restrict__ w3, const float* __restrict__ b3,
    float* __restrict__ out) {
    __shared__ float p[64], x1[128], x2[64];
    const int g = blockIdx.x, t = threadIdx.x;
    if (t < 64) {
        float cnt = cnts[g];
        if (cnt < 1.0f) cnt = 1.0f;
        p[t] = pooled[(size_t)g * HDIM + t] / cnt;
    }
    __syncthreads();
    float a = b1[t];
#pragma unroll 8
    for (int k = 0; k < 64; ++k) a += p[k] * w1[k * 128 + t];
    x1[t] = softplusf(a);
    __syncthreads();
    if (t < 64) {
        float a2 = b2[t];
#pragma unroll 8
        for (int k = 0; k < 128; ++k) a2 += x1[k] * w2[k * 64 + t];
        x2[t] = softplusf(a2);
    }
    __syncthreads();
    if (t == 0) {
        float s = b3[0];
        for (int k = 0; k < 64; ++k) s += x2[k] * w3[k];
        out[g] = s;
    }
}

// ---------------------------------------------------------------------------
static inline size_t align256(size_t x) { return (x + 255) & ~(size_t)255; }

extern "C" void kernel_launch(void* const* d_in, const int* in_sizes, int n_in,
                              void* d_out, int out_size, void* d_ws, size_t ws_size,
                              hipStream_t stream) {
    const float* node_features = (const float*)d_in[0];
    const int*   edge_index    = (const int*)d_in[1];
    const float* edge_features = (const float*)d_in[2];
    const int*   graph_index   = (const int*)d_in[3];
    const float* embed_w = (const float*)d_in[4];
    const float* embed_b = (const float*)d_in[5];
    const float* gate_w  = (const float*)d_in[6];
    const float* gate_b  = (const float*)d_in[7];
    const float* cand_w  = (const float*)d_in[8];
    const float* cand_b  = (const float*)d_in[9];
    const float* bn_g    = (const float*)d_in[10];
    const float* bn_b    = (const float*)d_in[11];
    const float* w1 = (const float*)d_in[12];
    const float* b1 = (const float*)d_in[13];
    const float* w2 = (const float*)d_in[14];
    const float* b2 = (const float*)d_in[15];
    const float* w3 = (const float*)d_in[16];
    const float* b3 = (const float*)d_in[17];

    const int Nn = in_sizes[0] / 35;       // 100000
    const int E  = in_sizes[2] / 10;       // 1600000
    const int Bg = 256;
    const int IN_DIM = 138;

    const int* srcp = edge_index;
    const int* dstp = edge_index + E;

    // workspace layout (agg has one extra dummy row for invalid-edge scatter)
    char* ws = (char*)d_ws;
    size_t off = 0;
    float*    h      = (float*)(ws + off);    off += align256((size_t)Nn * HDIM * 4);
    _Float16* h16    = (_Float16*)(ws + off); off += align256((size_t)Nn * HDIM * 2);
    float*    agg    = (float*)(ws + off);    off += align256((size_t)(Nn + 1) * HDIM * 4);
    float*    stats  = (float*)(ws + off);    off += align256(128 * 4);
    _Float16* pg     = (_Float16*)(ws + off); off += align256(NCHUNK * 4 * 32 * 16 * 2);
    _Float16* pc     = (_Float16*)(ws + off); off += align256(NCHUNK * 4 * 32 * 16 * 2);
    float*    pooled = (float*)(ws + off);    off += (size_t)Bg * HDIM * 4;
    float*    cnts   = (float*)(ws + off);    off += (size_t)Bg * 4;
    (void)ws_size; (void)n_in; (void)out_size;

    const int elemBlocks = (Nn * HDIM + 255) / 256;

    embed_kernel<<<elemBlocks, 256, 0, stream>>>(node_features, embed_w, embed_b,
                                                 h, h16, Nn);

    for (int layer = 0; layer < 3; ++layer) {
        hipMemsetAsync(agg, 0, (size_t)(Nn + 1) * HDIM * 4, stream);
        hipMemsetAsync(stats, 0, 128 * 4, stream);
        pack_weights<<<(2 * NCHUNK * 4 * 32 * 16 + 255) / 256, 256, 0, stream>>>(
            gate_w + (size_t)layer * IN_DIM * HDIM,
            cand_w + (size_t)layer * IN_DIM * HDIM, pg, pc);
        edge_conv<<<1024, 256, 0, stream>>>(h16, srcp, dstp, edge_features, pg, pc,
                                            gate_b + layer * HDIM,
                                            cand_b + layer * HDIM, agg, E, Nn);
        bn_stats<<<1024, 256, 0, stream>>>(agg, stats, Nn);
        bn_update<<<elemBlocks, 256, 0, stream>>>(h, h16, agg, stats,
                                                  bn_g + layer * HDIM,
                                                  bn_b + layer * HDIM, Nn);
    }

    hipMemsetAsync(pooled, 0, ((size_t)Bg * HDIM + Bg) * 4, stream); // pooled + cnts
    pool_kernel<<<elemBlocks, 256, 0, stream>>>(h, graph_index, pooled, cnts, Nn);
    mlp_kernel<<<Bg, 128, 0, stream>>>(pooled, cnts, w1, b1, w2, b2, w3, b3,
                                       (float*)d_out);
}
